// UFG_OT_Loss_57801669869629
// MI455X (gfx1250) — matrix operand order repaired and checked by reading, not compiled, not run on hardware
//
#include <hip/hip_runtime.h>
#include <hip/hip_bf16.h>
#include <math.h>

// ---------------------------------------------------------------------------
// UFG-OT loss for MI455X (gfx1250, wave32, WMMA).
// GEMMs: wave-level 64x64 tiles via v_wmma_f32_16x16x32_{f16,bf16}.
// ---------------------------------------------------------------------------

typedef __attribute__((ext_vector_type(16))) _Float16 v16h;
typedef __attribute__((ext_vector_type(8)))  _Float16 v8h;
typedef __attribute__((ext_vector_type(16))) __bf16   v16bf;
typedef __attribute__((ext_vector_type(8)))  __bf16   v8bf;
typedef __attribute__((ext_vector_type(8)))  float    v8f;

static constexpr int   NN   = 2048;
static constexpr int   DD   = 512;
static constexpr float TAU  = 0.01f;
static constexpr float EPSC = 1e-5f;
static constexpr float AA   = 0.01f;   // A0..A3 all 0.01

template<typename T> struct WT;
template<> struct WT<_Float16> {
  using V16 = v16h; using V8 = v8h;
  static __device__ inline v8f mma(V16 a, V16 b, v8f c) {
    return __builtin_amdgcn_wmma_f32_16x16x32_f16(false, a, false, b, (short)0, c, false, false);
  }
};
template<> struct WT<__bf16> {
  using V16 = v16bf; using V8 = v8bf;
  static __device__ inline v8f mma(V16 a, V16 b, v8f c) {
    return __builtin_amdgcn_wmma_f32_16x16x32_bf16(false, a, false, b, (short)0, c, false, false);
  }
};

// ---------------------------------------------------------------------------
// NT GEMM: D = A(row-major MxK) * B(row-major NxK)^T, f32 accumulate.
// MODE 0: out f32  = 1 - D            (x = 1 - img.txt^T)
// MODE 1: out bf16 = 1 - D            (c1/c2, symmetric cosine costs)
// MODE 2: out bf16 = D                (tmp = c2 @ t, via c2 * (t^T)^T)
// MODE 3: fused epilogue: g = D; cost = -x - a0*g - tau*log(t+eps);
//         y = -cost/a11.  Writes cost and y.
// Wave computes 64x64 (4x4 tiles of 16x16). Block = 8 waves (4x2 of 64x64) -> 256 rows x 128 cols.
// ---------------------------------------------------------------------------
template<typename T, int MODE>
__global__ __launch_bounds__(256) void k_gemm_nt(
    const T* __restrict__ A, const T* __restrict__ B, int K,
    void* __restrict__ outp,
    const float* __restrict__ xp, const float* tp,
    float* costp, float* yp,
    float a0, float tau, float inva11, float eps)
{
  using Tr  = WT<T>;
  using V16 = typename Tr::V16;
  using V8  = typename Tr::V8;
  union U { V16 v; V8 h[2]; };

  const int tid   = threadIdx.x;
  const int lane  = tid & 31;
  const int wave  = tid >> 5;
  const int m0    = blockIdx.y * 128 + (wave >> 1) * 64;
  const int n0    = blockIdx.x * 128 + (wave & 1) * 64;
  const int lrow  = lane & 15;   // row (A) / col (B,C) within 16-tile
  const int lhalf = lane >> 4;   // which K-half / M-half per ISA layout

  v8f acc[4][4] = {};

  for (int kb = 0; kb < K; kb += 32) {
    V16 afrag[4];
#pragma unroll
    for (int i = 0; i < 4; ++i) {
      // A 16x32 f16/bf16 layout: lane<16 holds K[0..7],K[16..23]; lane>=16 holds K[8..15],K[24..31]
      const T* p = A + (size_t)(m0 + 16 * i + lrow) * (size_t)K + kb + lhalf * 8;
      U u;
      u.h[0] = *(const V8*)(p);
      u.h[1] = *(const V8*)(p + 16);
      afrag[i] = u.v;
    }
#pragma unroll
    for (int j = 0; j < 4; ++j) {
      // B 32x16 layout: lane<16 holds K[0..7],K[16..23]; lane>=16 holds K[8..15],K[24..31] (same as A)
      const T* p = B + (size_t)(n0 + 16 * j + lrow) * (size_t)K + kb + lhalf * 8;
      U u;
      u.h[0] = *(const V8*)(p);
      u.h[1] = *(const V8*)(p + 16);
      V16 bfrag = u.v;
#pragma unroll
      for (int i = 0; i < 4; ++i)
        acc[i][j] = Tr::mma(afrag[i], bfrag, acc[i][j]);
    }
  }

  // C/D f32 16x16: lane<16 -> M = v, lane>=16 -> M = 8+v; N = lane&15
#pragma unroll
  for (int i = 0; i < 4; ++i) {
#pragma unroll
    for (int j = 0; j < 4; ++j) {
#pragma unroll
      for (int v = 0; v < 8; ++v) {
        const int grow = m0 + 16 * i + v + 8 * lhalf;
        const int gcol = n0 + 16 * j + lrow;
        const size_t idx = (size_t)grow * NN + gcol;
        const float val = acc[i][j][v];
        if constexpr (MODE == 0) {
          ((float*)outp)[idx] = 1.0f - val;
        } else if constexpr (MODE == 1) {
          ((__bf16*)outp)[idx] = (__bf16)(1.0f - val);
        } else if constexpr (MODE == 2) {
          ((__bf16*)outp)[idx] = (__bf16)val;
        } else {
          const float x  = xp[idx];
          const float tv = tp[idx];
          const float c  = -x - a0 * val - tau * __logf(tv + eps);
          costp[idx] = c;
          yp[idx]    = -c * inva11;
        }
      }
    }
  }
}

// float -> f16 feature conversion
__global__ void k_f2h(const float* __restrict__ in, _Float16* __restrict__ out, int n) {
  int i = blockIdx.x * blockDim.x + threadIdx.x;
  if (i < n) out[i] = (_Float16)in[i];
}

// init: t = 1/bs^2 (f32), t^T in bf16
__global__ void k_init_t(float* __restrict__ yt, __bf16* __restrict__ ttb, float v) {
  size_t i = (size_t)blockIdx.x * blockDim.x + threadIdx.x;
  yt[i]  = v;
  ttb[i] = (__bf16)v;
}

// row logsumexp of y -> a[i] = cA*(2*lp0 - lse_j(y[i,:]))
__global__ __launch_bounds__(256) void k_row_lse(const float* __restrict__ y,
                                                 float* __restrict__ a,
                                                 float cA, float lp0) {
  const int i = blockIdx.x;
  const float* row = y + (size_t)i * NN;
  float m = -INFINITY, s = 0.0f;
  for (int j = threadIdx.x; j < NN; j += 256) {
    const float v = row[j];
    if (v <= m) s += __expf(v - m);
    else        { s = s * __expf(m - v) + 1.0f; m = v; }
  }
  __shared__ float sm[256], ss[256];
  sm[threadIdx.x] = m; ss[threadIdx.x] = s;
  __syncthreads();
  for (int o = 128; o > 0; o >>= 1) {
    if ((int)threadIdx.x < o) {
      const float m2 = sm[threadIdx.x + o], s2 = ss[threadIdx.x + o];
      const float m1 = sm[threadIdx.x],     s1 = ss[threadIdx.x];
      if (m1 >= m2) { ss[threadIdx.x] = s1 + s2 * __expf(m2 - m1); }
      else          { sm[threadIdx.x] = m2; ss[threadIdx.x] = s2 + s1 * __expf(m1 - m2); }
    }
    __syncthreads();
  }
  if (threadIdx.x == 0) a[i] = cA * (2.0f * lp0 - (sm[0] + __logf(ss[0])));
}

// column logsumexp of y -> b[j] = cB*(2*lq0 - lse_i(y[:,j])).  blockDim (64,4)
__global__ __launch_bounds__(256) void k_col_lse(const float* __restrict__ y,
                                                 float* __restrict__ b,
                                                 float cB, float lq0) {
  const int col = blockIdx.x * 64 + threadIdx.x;
  const int ty  = threadIdx.y;
  float m = -INFINITY, s = 0.0f;
  for (int r = ty; r < NN; r += 4) {
    const float v = y[(size_t)r * NN + col];
    if (v <= m) s += __expf(v - m);
    else        { s = s * __expf(m - v) + 1.0f; m = v; }
  }
  __shared__ float sm[4][64], ss[4][64];
  sm[ty][threadIdx.x] = m; ss[ty][threadIdx.x] = s;
  __syncthreads();
  if (ty == 0) {
    for (int q = 1; q < 4; ++q) {
      const float m2 = sm[q][threadIdx.x], s2 = ss[q][threadIdx.x];
      if (m >= m2) { s = s + s2 * __expf(m2 - m); }
      else         { s = s2 + s * __expf(m - m2); m = m2; }
    }
    b[col] = cB * (2.0f * lq0 - (m + __logf(s)));
  }
}

// y = -cost/a11 + a[i] + b[j]
__global__ void k_update_y(const float* __restrict__ cost, float* __restrict__ y,
                           const float* __restrict__ a, const float* __restrict__ b,
                           float inva11) {
  const size_t idx = (size_t)blockIdx.x * blockDim.x + threadIdx.x;
  const int i = (int)(idx >> 11), j = (int)(idx & (NN - 1));
  y[idx] = -cost[idx] * inva11 + a[i] + b[j];
}

// t = exp(y) in place; also t^T in bf16 for the next GW GEMM
__global__ void k_exp_t(float* __restrict__ yt, __bf16* __restrict__ ttb) {
  const size_t idx = (size_t)blockIdx.x * blockDim.x + threadIdx.x;
  const int i = (int)(idx >> 11), j = (int)(idx & (NN - 1));
  const float t = __expf(fminf(yt[idx], 80.0f));
  yt[idx] = t;
  ttb[(size_t)j * NN + i] = (__bf16)t;
}

// per-row CE: rowloss[i] = lse_j(t[i,:]) - t[i, labels[i]]
__global__ __launch_bounds__(256) void k_loss_rows(const float* __restrict__ t,
                                                   const int* __restrict__ labels,
                                                   float* __restrict__ rl) {
  const int i = blockIdx.x;
  const float* row = t + (size_t)i * NN;
  float m = -INFINITY, s = 0.0f;
  for (int j = threadIdx.x; j < NN; j += 256) {
    const float v = row[j];
    if (v <= m) s += __expf(v - m);
    else        { s = s * __expf(m - v) + 1.0f; m = v; }
  }
  __shared__ float sm[256], ss[256];
  sm[threadIdx.x] = m; ss[threadIdx.x] = s;
  __syncthreads();
  for (int o = 128; o > 0; o >>= 1) {
    if ((int)threadIdx.x < o) {
      const float m2 = sm[threadIdx.x + o], s2 = ss[threadIdx.x + o];
      const float m1 = sm[threadIdx.x],     s1 = ss[threadIdx.x];
      if (m1 >= m2) { ss[threadIdx.x] = s1 + s2 * __expf(m2 - m1); }
      else          { sm[threadIdx.x] = m2; ss[threadIdx.x] = s2 + s1 * __expf(m1 - m2); }
    }
    __syncthreads();
  }
  if (threadIdx.x == 0)
    rl[i] = (sm[0] + __logf(ss[0])) - row[labels[i]];
}

__global__ __launch_bounds__(256) void k_loss_final(const float* __restrict__ rl,
                                                    float* __restrict__ out) {
  __shared__ float sb[256];
  float s = 0.0f;
  for (int k = threadIdx.x; k < NN; k += 256) s += rl[k];
  sb[threadIdx.x] = s;
  __syncthreads();
  for (int o = 128; o > 0; o >>= 1) {
    if ((int)threadIdx.x < o) sb[threadIdx.x] += sb[threadIdx.x + o];
    __syncthreads();
  }
  if (threadIdx.x == 0) out[0] = sb[0] * (1.0f / (float)NN);
}

extern "C" void kernel_launch(void* const* d_in, const int* in_sizes, int n_in,
                              void* d_out, int out_size, void* d_ws, size_t ws_size,
                              hipStream_t stream) {
  const float* img = (const float*)d_in[0];
  const float* txt = (const float*)d_in[1];
  const int*   lab = (const int*)d_in[2];
  float* out = (float*)d_out;

  // workspace carve-up (all offsets 256B-aligned)
  char* w = (char*)d_ws;
  size_t off = 0;
  auto carve = [&](size_t bytes) { void* p = w + off; off += (bytes + 255) & ~(size_t)255; return p; };
  _Float16* imgh = (_Float16*)carve((size_t)NN * DD * 2);
  _Float16* txth = (_Float16*)carve((size_t)NN * DD * 2);
  float*    X    = (float*)   carve((size_t)NN * NN * 4);
  float*    COST = (float*)   carve((size_t)NN * NN * 4);
  float*    YT   = (float*)   carve((size_t)NN * NN * 4);  // holds t, then y, then t
  __bf16*   C1B  = (__bf16*)  carve((size_t)NN * NN * 2);
  __bf16*   C2B  = (__bf16*)  carve((size_t)NN * NN * 2);
  __bf16*   TTB  = (__bf16*)  carve((size_t)NN * NN * 2);  // t^T bf16
  __bf16*   TMPB = (__bf16*)  carve((size_t)NN * NN * 2);  // c2 @ t, bf16
  float*    AV   = (float*)   carve((size_t)NN * 4);
  float*    BV   = (float*)   carve((size_t)NN * 4);
  float*    RL   = (float*)   carve((size_t)NN * 4);

  const float a11    = AA + TAU;            // 0.02
  const float inva11 = 1.0f / a11;
  const float cA     = AA / (AA + a11);     // 1/3
  const float lp0    = logf(1.0f / (float)NN + EPSC);
  const float invbs2 = (1.0f / (float)NN) * (1.0f / (float)NN);

  const dim3 gemmGrid(NN / 128, NN / 128);  // 16x16
  const dim3 blk256(256);
  const int  nFeat = NN * DD;
  const int  nMat  = NN * NN;

  // features -> f16
  k_f2h<<<nFeat / 256, blk256, 0, stream>>>(img, imgh, nFeat);
  k_f2h<<<nFeat / 256, blk256, 0, stream>>>(txt, txth, nFeat);

  // x = 1 - img.txt^T (f32); c1/c2 = 1 - gram (bf16)
  k_gemm_nt<_Float16, 0><<<gemmGrid, blk256, 0, stream>>>(imgh, txth, DD, X,
      nullptr, nullptr, nullptr, nullptr, 0.f, 0.f, 0.f, 0.f);
  k_gemm_nt<_Float16, 1><<<gemmGrid, blk256, 0, stream>>>(imgh, imgh, DD, C1B,
      nullptr, nullptr, nullptr, nullptr, 0.f, 0.f, 0.f, 0.f);
  k_gemm_nt<_Float16, 1><<<gemmGrid, blk256, 0, stream>>>(txth, txth, DD, C2B,
      nullptr, nullptr, nullptr, nullptr, 0.f, 0.f, 0.f, 0.f);

  // t0 = 1/bs^2
  k_init_t<<<nMat / 256, blk256, 0, stream>>>(YT, TTB, invbs2);

  for (int m = 0; m < 4; ++m) {
    // tmp = c2 @ t  ==  c2 * (t^T)^T   (bf16 out)
    k_gemm_nt<__bf16, 2><<<gemmGrid, blk256, 0, stream>>>(C2B, TTB, NN, TMPB,
        nullptr, nullptr, nullptr, nullptr, 0.f, 0.f, 0.f, 0.f);
    // g = tmp @ c1 == tmp * c1^T (c1 symmetric); fused cost & y0 epilogue
    k_gemm_nt<__bf16, 3><<<gemmGrid, blk256, 0, stream>>>(TMPB, C1B, NN, nullptr,
        X, YT, COST, YT, AA, TAU, inva11, EPSC);
    // Sinkhorn inner iterations
    for (int it = 0; it < 4; ++it) {
      k_row_lse<<<NN, blk256, 0, stream>>>(YT, AV, cA, lp0);
      k_col_lse<<<NN / 64, dim3(64, 4), 0, stream>>>(YT, BV, cA, lp0);
      k_update_y<<<nMat / 256, blk256, 0, stream>>>(COST, YT, AV, BV, inva11);
    }
    // t = exp(y); also t^T bf16 for next iteration
    k_exp_t<<<nMat / 256, blk256, 0, stream>>>(YT, TTB);
  }

  // loss = mean_i( lse_j t[i,:] - t[i, labels[i]] )
  k_loss_rows<<<NN, blk256, 0, stream>>>(YT, lab, RL);
  k_loss_final<<<1, blk256, 0, stream>>>(RL, out);
}
